// MultiheadAttention_26396869001773
// MI455X (gfx1250) — matrix operand order repaired and checked by reading, not compiled, not run on hardware
//
#include <hip/hip_runtime.h>

#define Bn 8
#define Sn 2048
#define Dn 512
#define Hn 8
#define DEPTHn 64

typedef _Float16 v16h __attribute__((ext_vector_type(16)));
typedef _Float16 v8h  __attribute__((ext_vector_type(8)));
typedef float    v8f  __attribute__((ext_vector_type(8)));
typedef float    v4f  __attribute__((ext_vector_type(4)));

#define WMMA_F16(a, b, c) \
  __builtin_amdgcn_wmma_f32_16x16x32_f16(false, (a), false, (b), (short)0, (c), false, false)

// ---- fragment loaders -------------------------------------------------------
// A fragment in this code's contiguous k order (not AMD's table order): lane<16 holds row M=lane, K=kb+0..15;
// lane>=16 holds row M=lane-16, K=kb+16..31.
static __device__ __forceinline__ v16h a_frag_f32(const float* __restrict__ rowp,
                                                  int kb, int half) {
  const float* p0 = rowp + kb + half * 16;
  const float* p1 = p0 + 8;
  v4f x0 = *(const v4f*)(p0);
  v4f x1 = *(const v4f*)(p0 + 4);
  v4f y0 = *(const v4f*)(p1);
  v4f y1 = *(const v4f*)(p1 + 4);
  v16h a;
#pragma unroll
  for (int j = 0; j < 4; ++j) {
    a[j]      = (_Float16)x0[j];
    a[j + 4]  = (_Float16)x1[j];
    a[j + 8]  = (_Float16)y0[j];
    a[j + 12] = (_Float16)y1[j];
  }
  return a;
}

static __device__ __forceinline__ v16h a_frag_f16(const _Float16* rowp,
                                                  int kb, int half) {
  v8h lo = *(const v8h*)(rowp + kb + half * 16);
  v8h hi = *(const v8h*)(rowp + kb + 8 + half * 16);
  v16h a;
#pragma unroll
  for (int j = 0; j < 8; ++j) { a[j] = lo[j]; a[j + 8] = hi[j]; }
  return a;
}

// B fragment in the same contiguous k order as A: lane n (0-15) holds column n, K=kb+0..15;
// lane n+16 holds column n, K=kb+16..31.  rowp points at the row of the
// transposed source (B[k][n] = rowp[k]), so loads are contiguous.
static __device__ __forceinline__ v16h b_frag_f32(const float* __restrict__ rowp,
                                                  int kb, int half) {
  const float* p = rowp + kb + half * 16;
  v4f x0 = *(const v4f*)(p);
  v4f x1 = *(const v4f*)(p + 4);
  v4f x2 = *(const v4f*)(p + 8);
  v4f x3 = *(const v4f*)(p + 12);
  v16h b;
#pragma unroll
  for (int j = 0; j < 4; ++j) {
    b[j]      = (_Float16)x0[j];
    b[j + 4]  = (_Float16)x1[j];
    b[j + 8]  = (_Float16)x2[j];
    b[j + 12] = (_Float16)x3[j];
  }
  return b;
}

static __device__ __forceinline__ v16h b_frag_f16(const _Float16* rowp,
                                                  int kb, int half) {
  return *(const v16h*)(rowp + kb + half * 16);
}

// ---- kernel 1: Q/K/V projection  Y = X @ W^T + b  -> f16, head-split layout -
// Two independent accumulator chains hide the WMMA->WMMA RAW hazard.
__global__ __launch_bounds__(256) void proj_qkv_kernel(
    const float* __restrict__ X,     // [B*S, 512]
    const float* __restrict__ W,     // [512, 512]
    const float* __restrict__ bias,  // [512]
    _Float16* __restrict__ Y,        // head layout: base + s*ss + d*sd
    int ss, int sd) {
  const int lane = threadIdx.x & 31;
  const int wave = threadIdx.x >> 5;
  const int half = lane >> 4;
  const int l16  = lane & 15;
  const int mbase = blockIdx.x * 16;
  const int nbase = blockIdx.y * 128 + wave * 16;
  const float* Arow = X + (size_t)(mbase + l16) * Dn;
  const float* Brow = W + (size_t)(nbase + l16) * Dn;
  v8f acc0 = {}, acc1 = {};
#pragma unroll 2
  for (int kb = 0; kb < Dn; kb += 64) {
    v16h aA = a_frag_f32(Arow, kb, half);
    v16h bA = b_frag_f32(Brow, kb, half);
    v16h aB = a_frag_f32(Arow, kb + 32, half);
    v16h bB = b_frag_f32(Brow, kb + 32, half);
    acc0 = WMMA_F16(aA, bA, acc0);
    acc1 = WMMA_F16(aB, bB, acc1);
  }
  const v8f acc = acc0 + acc1;
  const int n  = nbase + l16;
  const float bv = bias[n];
  const int h  = n >> 6;
  const int dd = n & 63;
  const int b0 = mbase >> 11;
  const size_t headbase = (size_t)(b0 * Hn + h) * (size_t)(Sn * DEPTHn);
#pragma unroll
  for (int i = 0; i < 8; ++i) {
    const int s = (mbase & (Sn - 1)) + half * 8 + i;
    Y[headbase + (size_t)s * ss + (size_t)dd * sd] = (_Float16)(acc[i] + bv);
  }
}

// ---- kernel 2 (FULLY FUSED): logits + mask + adjoin + softmax + attn write
//      + ctx = attn @ V, all in one pass over a 16-row q-tile.
// Wave w owns key-columns [w*256, (w+1)*256) as 16 WMMA tiles in registers.
// grid: (S/16, 1, B*H), block 256.
__global__ __launch_bounds__(256) void attn_ctx_fused_kernel(
    const _Float16* __restrict__ Qh,   // [B*H][S][64]
    const _Float16* __restrict__ Kh,   // [B*H][S][64]
    const _Float16* __restrict__ VhT,  // [B*H][64][S]
    const float* __restrict__ adjoin,  // [B][S][S]
    const float* __restrict__ mask,    // [B][S]
    float* __restrict__ attn,          // [B*H][S][S]  (normalized, output)
    float* __restrict__ ctx)           // [B][S][512]
{
  // LDS: row-major f16 staging of the normalized 16x2048 attn block
  // (+8 halves row padding -> row stride 1028 dwords, distinct banks per row),
  // 16x64 fp32 ctx reduction buffer, 8x16 softmax scratch.
  __shared__ _Float16 att_lds[16][2056];
  __shared__ float    cbuf[16][64];
  __shared__ float    red[8][16];

  const int lane = threadIdx.x & 31;
  const int wave = threadIdx.x >> 5;
  const int half = lane >> 4;
  const int l16  = lane & 15;
  const int bh = blockIdx.z;
  const int b0 = bh >> 3;
  const int h  = bh & 7;
  const int qbase = blockIdx.x * 16;
  const int cbase = wave * 256;

  // zero the ctx reduction buffer (separated from the atomics by 3 barriers)
  ((float4*)cbuf)[threadIdx.x] = make_float4(0.f, 0.f, 0.f, 0.f);

  // Q fragments (16 rows x 64 depth), reused across all 16 column tiles
  const _Float16* Arow = Qh + ((size_t)bh * Sn + qbase + l16) * DEPTHn;
  const v16h a0 = a_frag_f16(Arow, 0, half);
  const v16h a1 = a_frag_f16(Arow, 32, half);

  // 16 accumulator tiles: 16 q-rows x 256 columns held in 128 VGPRs/lane.
  // Tiles processed in pairs -> two independent WMMA chains per pair.
  v8f acc[16];
#pragma unroll
  for (int t = 0; t < 16; t += 2) {
    const _Float16* Brow0 = Kh + ((size_t)bh * Sn + cbase + t * 16 + l16) * DEPTHn;
    const _Float16* Brow1 = Brow0 + 16 * DEPTHn;
    v16h b00 = b_frag_f16(Brow0, 0, half);
    v16h b01 = b_frag_f16(Brow0, 32, half);
    v16h b10 = b_frag_f16(Brow1, 0, half);
    v16h b11 = b_frag_f16(Brow1, 32, half);
    v8f c0 = {}, c1 = {};
    c0 = WMMA_F16(a0, b00, c0);
    c1 = WMMA_F16(a0, b10, c1);
    c0 = WMMA_F16(a1, b01, c0);
    c1 = WMMA_F16(a1, b11, c1);
    acc[t] = c0;
    acc[t + 1] = c1;
  }

  // scale + mask + adjoin, track per-row running maxima (8 rows per lane)
  float mx[8];
#pragma unroll
  for (int i = 0; i < 8; ++i) mx[i] = -3.0e38f;
#pragma unroll
  for (int t = 0; t < 16; ++t) {
    const int n = cbase + t * 16 + l16;
    const float mv = mask[(size_t)b0 * Sn + n] * -1e9f;
#pragma unroll
    for (int i = 0; i < 8; ++i) {
      const int q = qbase + half * 8 + i;
      float val = acc[t][i] * 0.125f + mv + adjoin[((size_t)b0 * Sn + q) * Sn + n];
      acc[t][i] = val;
      mx[i] = fmaxf(mx[i], val);
    }
  }
  // reduce max over the 16 lanes of this half (they hold the same 8 rows)
#pragma unroll
  for (int off = 1; off < 16; off <<= 1) {
#pragma unroll
    for (int i = 0; i < 8; ++i) mx[i] = fmaxf(mx[i], __shfl_xor(mx[i], off, 32));
  }
  if (l16 == 0) {
#pragma unroll
    for (int i = 0; i < 8; ++i) red[wave][half * 8 + i] = mx[i];
  }
  __syncthreads();
  float gmx[8];
#pragma unroll
  for (int i = 0; i < 8; ++i) {
    float m = red[0][half * 8 + i];
#pragma unroll
    for (int w2 = 1; w2 < 8; ++w2) m = fmaxf(m, red[w2][half * 8 + i]);
    gmx[i] = m;
  }
  __syncthreads();

  // exponentiate in registers, accumulate per-row partial sums
  float sm[8];
#pragma unroll
  for (int i = 0; i < 8; ++i) sm[i] = 0.f;
#pragma unroll
  for (int t = 0; t < 16; ++t) {
#pragma unroll
    for (int i = 0; i < 8; ++i) {
      float e = __expf(acc[t][i] - gmx[i]);
      acc[t][i] = e;
      sm[i] += e;
    }
  }
#pragma unroll
  for (int off = 1; off < 16; off <<= 1) {
#pragma unroll
    for (int i = 0; i < 8; ++i) sm[i] += __shfl_xor(sm[i], off, 32);
  }
  if (l16 == 0) {
#pragma unroll
    for (int i = 0; i < 8; ++i) red[wave][half * 8 + i] = sm[i];
  }
  __syncthreads();
  float inv[8];
#pragma unroll
  for (int i = 0; i < 8; ++i) {
    float s = red[0][half * 8 + i];
#pragma unroll
    for (int w2 = 1; w2 < 8; ++w2) s += red[w2][half * 8 + i];
    inv[i] = 1.0f / s;
  }

  // normalize: non-temporal fp32 global write of attn (write-once stream;
  // keep adjoin's L2 working set resident) + f16 staging into LDS.
  // This wave re-reads only its own staged columns -> same-wave DS ordering,
  // no cross-wave barrier needed before the ctx phase.
#pragma unroll
  for (int t = 0; t < 16; ++t) {
    const int n = cbase + t * 16 + l16;
#pragma unroll
    for (int i = 0; i < 8; ++i) {
      const int q = qbase + half * 8 + i;
      const float p = acc[t][i] * inv[i];
      __builtin_nontemporal_store(p, &attn[((size_t)bh * Sn + q) * Sn + n]);
      att_lds[half * 8 + i][n] = (_Float16)p;
    }
  }

  // ---- ctx phase: this wave's K-range partial of attn @ V -------------------
  // A-frags from LDS (row-major, lane = q-row), B-frags from VhT (contiguous).
  // 4 independent accumulator chains -> no WMMA RAW stalls.
  v8f cacc[4];
#pragma unroll
  for (int nt = 0; nt < 4; ++nt) cacc[nt] = (v8f){};
  const _Float16* lrow = &att_lds[l16][0];
#pragma unroll
  for (int kb = 0; kb < 256; kb += 32) {
    v16h a = a_frag_f16(lrow, cbase + kb, half);
#pragma unroll
    for (int nt = 0; nt < 4; ++nt) {
      const _Float16* Brow = VhT + ((size_t)bh * DEPTHn + nt * 16 + l16) * Sn;
      v16h b = b_frag_f16(Brow, cbase + kb, half);
      cacc[nt] = WMMA_F16(a, b, cacc[nt]);
    }
  }
  // cross-wave K reduction into LDS (ds_add_f32)
#pragma unroll
  for (int nt = 0; nt < 4; ++nt) {
    const int dd = nt * 16 + l16;
#pragma unroll
    for (int i = 0; i < 8; ++i) {
      atomicAdd(&cbuf[half * 8 + i][dd], cacc[nt][i]);
    }
  }
  __syncthreads();

  // coalesced float4 writeout of ctx [16 rows x 64 depth]
  {
    const int row = threadIdx.x >> 4;         // 0..15
    const int dd4 = (threadIdx.x & 15) * 4;   // 0..60
    const float4 vv = *(const float4*)&cbuf[row][dd4];
    float* dst = &ctx[((size_t)b0 * Sn + qbase + row) * Dn + h * DEPTHn + dd4];
    *(float4*)dst = vv;
  }
}

// ---- kernel 3: out = ctx @ dense_w^T + dense_b  (fp32 out, NT stores) -------
__global__ __launch_bounds__(256) void dense_kernel(
    const float* __restrict__ X,     // [B*S, 512] (ctx)
    const float* __restrict__ W,     // [512, 512]
    const float* __restrict__ bias,  // [512]
    float* __restrict__ Y)           // [B*S, 512]
{
  const int lane = threadIdx.x & 31;
  const int wave = threadIdx.x >> 5;
  const int half = lane >> 4;
  const int l16  = lane & 15;
  const int mbase = blockIdx.x * 16;
  const int nbase = blockIdx.y * 128 + wave * 16;
  const float* Arow = X + (size_t)(mbase + l16) * Dn;
  const float* Brow = W + (size_t)(nbase + l16) * Dn;
  v8f acc0 = {}, acc1 = {};
#pragma unroll 2
  for (int kb = 0; kb < Dn; kb += 64) {
    v16h aA = a_frag_f32(Arow, kb, half);
    v16h bA = b_frag_f32(Brow, kb, half);
    v16h aB = a_frag_f32(Arow, kb + 32, half);
    v16h bB = b_frag_f32(Brow, kb + 32, half);
    acc0 = WMMA_F16(aA, bA, acc0);
    acc1 = WMMA_F16(aB, bB, acc1);
  }
  const v8f acc = acc0 + acc1;
  const int n = nbase + l16;
  const float bv = bias[n];
#pragma unroll
  for (int i = 0; i < 8; ++i) {
    __builtin_nontemporal_store(acc[i] + bv,
                                &Y[(size_t)(mbase + half * 8 + i) * Dn + n]);
  }
}

// ---- host-side launcher -----------------------------------------------------
extern "C" void kernel_launch(void* const* d_in, const int* in_sizes, int n_in,
                              void* d_out, int out_size, void* d_ws, size_t ws_size,
                              hipStream_t stream) {
  const float* q      = (const float*)d_in[0];
  const float* k      = (const float*)d_in[1];
  const float* v      = (const float*)d_in[2];
  const float* mask   = (const float*)d_in[3];
  const float* adjoin = (const float*)d_in[4];
  const float* wq_w   = (const float*)d_in[5];
  const float* wq_b   = (const float*)d_in[6];
  const float* wk_w   = (const float*)d_in[7];
  const float* wk_b   = (const float*)d_in[8];
  const float* wv_w   = (const float*)d_in[9];
  const float* wv_b   = (const float*)d_in[10];
  const float* dw     = (const float*)d_in[11];
  const float* db     = (const float*)d_in[12];

  const size_t headElems = (size_t)Bn * Hn * Sn * DEPTHn;  // 8,388,608
  _Float16* Qh  = (_Float16*)d_ws;
  _Float16* Kh  = Qh + headElems;
  _Float16* VhT = Kh + headElems;
  float*    ctx = (float*)(VhT + headElems);               // [B*S, 512] fp32

  float* out  = (float*)d_out;                    // [B,S,D]
  float* attn = out + (size_t)Bn * Sn * Dn;       // [B,H,S,S]

  dim3 gproj(Bn * Sn / 16, Dn / 128, 1);
  proj_qkv_kernel<<<gproj, 256, 0, stream>>>(q, wq_w, wq_b, Qh, DEPTHn, 1);
  proj_qkv_kernel<<<gproj, 256, 0, stream>>>(k, wk_w, wk_b, Kh, DEPTHn, 1);
  proj_qkv_kernel<<<gproj, 256, 0, stream>>>(v, wv_w, wv_b, VhT, 1, Sn);

  dim3 gattn(Sn / 16, 1, Bn * Hn);
  attn_ctx_fused_kernel<<<gattn, 256, 0, stream>>>(Qh, Kh, VhT, adjoin, mask,
                                                   attn, ctx);

  dense_kernel<<<gproj, 256, 0, stream>>>(ctx, dw, db, out);
}
